// HierarchicalEncoder_48842368090159
// MI455X (gfx1250) — matrix operand order repaired and checked by reading, not compiled, not run on hardware
//
#include <hip/hip_runtime.h>
#include <hip/hip_bf16.h>

typedef __attribute__((ext_vector_type(16))) _Float16 v16h;
typedef __attribute__((ext_vector_type(8)))  _Float16 v8h;
typedef __attribute__((ext_vector_type(8)))  float    v8f;
typedef __attribute__((ext_vector_type(4)))  unsigned v4u;
typedef __attribute__((ext_vector_type(8)))  int      v8i;
typedef __attribute__((ext_vector_type(4)))  int      v4i;

#define HEADS 8
#define DH 32
#define DMODEL 256
#define INV_SQRT_DH 0.17677669529663687f

// ---------------------------------------------------------------------------
// helpers
// ---------------------------------------------------------------------------
__device__ __forceinline__ float gelu_tanh(float x) {
    const float c0 = 0.7978845608028654f;   // sqrt(2/pi)
    float x3 = x * x * x;
    return 0.5f * x * (1.0f + tanhf(c0 * (x + 0.044715f * x3)));
}

// monotonic float<->uint encoding so we can do float atomic-max via atomicMax(u32)
__device__ __forceinline__ unsigned fenc(float f) {
    unsigned u = __float_as_uint(f);
    return (u & 0x80000000u) ? ~u : (u | 0x80000000u);
}
__device__ __forceinline__ float fdec(unsigned e) {
    unsigned u = (e & 0x80000000u) ? (e ^ 0x80000000u) : ~e;
    return __uint_as_float(u);
}

// ---------------------------------------------------------------------------
// WMMA GEMM: C[M,256] = A[M,K] @ W[K,256] + bias   (fp32 in/out, f16 WMMA)
//
// Block = 128 threads (4 waves) computing a 16(M) x 256(N) stripe.
//  - A chunk 16x32 staged in LDS as f16 (cooperative float4 loads), each wave
//    builds its A fragment with two 16B ds reads matching the ISA 7.12.2
//    lane layout.
//  - B chunk 32x256 fp32 DMA'd global->LDS by the Tensor Data Mover
//    (tensor_load_to_lds, TENSORcnt) issued by wave 0 once per K-chunk.
//  - Each wave owns 64 N-columns -> 4 accumulators, 4 WMMAs per chunk
//    reusing its A fragment.
// ---------------------------------------------------------------------------
__global__ void __launch_bounds__(128)
k_gemm_bias(const float* __restrict__ A, const float* __restrict__ W,
            const float* __restrict__ bias, float* __restrict__ C,
            int M, int K) {
    __shared__ _Float16 sA[16 * 32];
    __shared__ float    sB[32 * 256];

    int tid = threadIdx.x;
    int wave = tid >> 5, lane = tid & 31;
    int lr = lane & 15, hi = lane >> 4;
    int m0 = blockIdx.x << 4;
    (void)M;

    v8f c0 = {}, c1 = {}, c2 = {}, c3 = {};

    unsigned ldsB = (unsigned)(unsigned long long)(&sB[0]);

    for (int k0 = 0; k0 < K; k0 += 32) {
        __syncthreads();   // previous chunk fully consumed before refill

        // ---- cooperative A tile load: 512 floats, 4 per thread, f32->f16 ----
        {
            int idx = tid * 4;
            int r = idx >> 5, cc = idx & 31;
            const float4 f = *(const float4*)(A + (size_t)(m0 + r) * (size_t)K + k0 + cc);
            sA[idx + 0] = (_Float16)f.x; sA[idx + 1] = (_Float16)f.y;
            sA[idx + 2] = (_Float16)f.z; sA[idx + 3] = (_Float16)f.w;
        }

        // ---- TDM: B chunk [32 x 256] fp32 -> LDS (D# per CDNA5 ISA ch.8) ----
        if (wave == 0) {
            unsigned long long ga = (unsigned long long)(W + (size_t)k0 * DMODEL);
            v4u g0; v8i g1; v4i g2 = {}, g3 = {}; v8i g4 = {};
            g0[0] = 1u;                                     // count=1, load, no gather
            g0[1] = ldsB;                                   // lds_addr
            g0[2] = (unsigned)(ga & 0xFFFFFFFFu);           // global_addr[31:0]
            g0[3] = (unsigned)((ga >> 32) & 0x01FFFFFFu)    // global_addr[56:32]
                    | (2u << 30);                           // type = 2 (image)
            g1[0] = (int)(2u << 16);                        // data_size = 4B
            g1[1] = (int)((unsigned)DMODEL << 16);          // tensor_dim0[15:0]=256
            g1[2] = (int)(((unsigned)(K - k0) & 0xFFFFu) << 16); // tensor_dim1[15:0]
            g1[3] = (int)((unsigned)DMODEL << 16);          // tile_dim0 = 256
            g1[4] = 32;                                     // tile_dim1 = 32
            g1[5] = DMODEL;                                 // tensor_dim0_stride = 256
            g1[6] = 0;
            g1[7] = 0;
            __builtin_amdgcn_tensor_load_to_lds(g0, g1, g2, g3, g4, 0);
            __builtin_amdgcn_s_wait_tensorcnt(0);
        }
        __syncthreads();   // LDS tiles visible to all waves

        // ---- A fragment from LDS: two 16B reads (ISA 7.12.2 16-bit A layout) ----
        v16h a;
        {
            const v8h a0 = *(const v8h*)&sA[lr * 32 + hi * 8];
            const v8h a1 = *(const v8h*)&sA[lr * 32 + 16 + hi * 8];
#pragma unroll
            for (int e = 0; e < 8; ++e) { a[e] = a0[e]; a[8 + e] = a1[e]; }
        }

        // ---- 4 B fragments + 4 WMMAs (wave covers cols wave*64 .. +63) ----
#pragma unroll
        for (int s = 0; s < 4; ++s) {
            int col = (wave << 6) + (s << 4) + lr;
            const float* bc = &sB[hi * 8 * DMODEL + col];
            v16h b;
#pragma unroll
            for (int e = 0; e < 16; ++e) b[e] = (_Float16)bc[(((e >> 3) << 4) + (e & 7)) * DMODEL];
            v8f& acc = (s == 0) ? c0 : (s == 1) ? c1 : (s == 2) ? c2 : c3;
            acc = __builtin_amdgcn_wmma_f32_16x16x32_f16(false, a, false, b,
                                                         (short)0, acc, false, false);
        }
    }

    // ---- store: lane<16: col, VGPR r -> row m0+r; lane>=16 -> row m0+8+r ----
#pragma unroll
    for (int s = 0; s < 4; ++s) {
        int col = (wave << 6) + (s << 4) + lr;
        float bv = bias ? bias[col] : 0.0f;
        const v8f& acc = (s == 0) ? c0 : (s == 1) ? c1 : (s == 2) ? c2 : c3;
        int rbase = m0 + hi * 8;
#pragma unroll
        for (int r = 0; r < 8; ++r)
            C[(size_t)(rbase + r) * DMODEL + col] = acc[r] + bv;
    }
}

// ---------------------------------------------------------------------------
// fill
// ---------------------------------------------------------------------------
__global__ void k_fill_u32(unsigned* p, unsigned v, int n) {
    int i = blockIdx.x * blockDim.x + threadIdx.x;
    if (i < n) p[i] = v;
}

// ---------------------------------------------------------------------------
// edge kernels: one thread per (edge, head); edge bias e = ea@We+be recomputed
// on the fly (32x32 dot per head) to avoid a 128MB materialized tensor.
// ---------------------------------------------------------------------------
__global__ void k_edge_logits(const float* __restrict__ q, const float* __restrict__ k,
                              const float* __restrict__ ea,
                              const float* __restrict__ We, const float* __restrict__ be,
                              const int* __restrict__ ei,
                              float* __restrict__ logits, int E) {
    int t = blockIdx.x * blockDim.x + threadIdx.x;
    if (t >= E * HEADS) return;
    int e = t >> 3, h = t & 7;
    int src = ei[e], dst = ei[E + e];
    const float* eav = ea + (size_t)e * 32;
    const float* qv = q + (size_t)dst * DMODEL + h * DH;
    const float* kv = k + (size_t)src * DMODEL + h * DH;
    float lg = 0.0f;
    for (int d = 0; d < DH; ++d) {
        float acc = be[h * DH + d];
        const float* wcol = We + h * DH + d;
#pragma unroll
        for (int j = 0; j < 32; ++j) acc += eav[j] * wcol[(size_t)j * DMODEL];
        lg += qv[d] * (kv[d] + acc);
    }
    logits[t] = lg * INV_SQRT_DH;
}

__global__ void k_seg_max(const float* __restrict__ logits, const int* __restrict__ ei,
                          unsigned* __restrict__ menc, int E) {
    int t = blockIdx.x * blockDim.x + threadIdx.x;
    if (t >= E * HEADS) return;
    int e = t >> 3, h = t & 7;
    int dst = ei[E + e];
    atomicMax(&menc[dst * HEADS + h], fenc(logits[t]));
}

__global__ void k_edge_exp(float* __restrict__ logits, const unsigned* __restrict__ menc,
                           const int* __restrict__ ei, float* __restrict__ sbuf, int E) {
    int t = blockIdx.x * blockDim.x + threadIdx.x;
    if (t >= E * HEADS) return;
    int e = t >> 3, h = t & 7;
    int dst = ei[E + e];
    float m = fdec(menc[dst * HEADS + h]);
    float ex = __expf(logits[t] - m);
    logits[t] = ex;
    atomicAdd(&sbuf[dst * HEADS + h], ex);
}

__global__ void k_edge_agg(const float* __restrict__ ex, const float* __restrict__ sbuf,
                           const float* __restrict__ v, const float* __restrict__ ea,
                           const float* __restrict__ We, const float* __restrict__ be,
                           const int* __restrict__ ei, float* __restrict__ agg, int E) {
    int t = blockIdx.x * blockDim.x + threadIdx.x;
    if (t >= E * HEADS) return;
    int e = t >> 3, h = t & 7;
    int src = ei[e], dst = ei[E + e];
    float alpha = ex[t] / (sbuf[dst * HEADS + h] + 1e-9f);
    const float* eav = ea + (size_t)e * 32;
    const float* vv = v + (size_t)src * DMODEL + h * DH;
    for (int d = 0; d < DH; ++d) {
        float acc = be[h * DH + d];
        const float* wcol = We + h * DH + d;
#pragma unroll
        for (int j = 0; j < 32; ++j) acc += eav[j] * wcol[(size_t)j * DMODEL];
        atomicAdd(&agg[(size_t)dst * DMODEL + h * DH + d], alpha * (vv[d] + acc));
    }
}

// ---------------------------------------------------------------------------
// LayerNorm (row of 256) + tanh-GELU, LDS tree reduction
// ---------------------------------------------------------------------------
__global__ void k_ln_gelu(const float* __restrict__ x, const float* __restrict__ g,
                          const float* __restrict__ b, float* __restrict__ out) {
    __shared__ float red[DMODEL];
    int row = blockIdx.x, c = threadIdx.x;
    float v = x[(size_t)row * DMODEL + c];
    red[c] = v; __syncthreads();
    for (int s = DMODEL / 2; s > 0; s >>= 1) {
        if (c < s) red[c] += red[c + s];
        __syncthreads();
    }
    float mean = red[0] * (1.0f / DMODEL);
    __syncthreads();
    float dv = v - mean;
    red[c] = dv * dv; __syncthreads();
    for (int s = DMODEL / 2; s > 0; s >>= 1) {
        if (c < s) red[c] += red[c + s];
        __syncthreads();
    }
    float var = red[0] * (1.0f / DMODEL);
    float y = dv * rsqrtf(var + 1e-5f) * g[c] + b[c];
    out[(size_t)row * DMODEL + c] = gelu_tanh(y);
}

// ---------------------------------------------------------------------------
// Cross-attention, online softmax (flash-style), one thread per (query, head)
// ---------------------------------------------------------------------------
__global__ void k_xattn(const float* __restrict__ q, const float* __restrict__ k,
                        const float* __restrict__ v, float* __restrict__ out,
                        int Nq, int Nk) {
    int t = blockIdx.x * blockDim.x + threadIdx.x;
    if (t >= Nq * HEADS) return;
    int qi = t >> 3, h = t & 7;
    float qv[DH], o[DH];
#pragma unroll
    for (int d = 0; d < DH; ++d) { qv[d] = q[(size_t)qi * DMODEL + h * DH + d]; o[d] = 0.0f; }
    float m = -1e30f, l = 0.0f;
    for (int kk = 0; kk < Nk; ++kk) {
        const float* kv = k + (size_t)kk * DMODEL + h * DH;
        if (kk + 8 < Nk) __builtin_prefetch(kv + 8 * DMODEL, 0, 3);
        float s = 0.0f;
#pragma unroll
        for (int d = 0; d < DH; ++d) s += qv[d] * kv[d];
        s *= INV_SQRT_DH;
        float nm = fmaxf(m, s);
        float corr = __expf(m - nm), p = __expf(s - nm);
        l = l * corr + p;
        const float* vv = v + (size_t)kk * DMODEL + h * DH;
#pragma unroll
        for (int d = 0; d < DH; ++d) o[d] = o[d] * corr + p * vv[d];
        m = nm;
    }
    float inv = 1.0f / l;
#pragma unroll
    for (int d = 0; d < DH; ++d) out[(size_t)qi * DMODEL + h * DH + d] = o[d] * inv;
}

// ---------------------------------------------------------------------------
// column means over N rows -> pooled[0..255] (coalesced: thread = column)
// ---------------------------------------------------------------------------
__global__ void k_mean_pool(const float* __restrict__ a, float* __restrict__ pooled, int N) {
    int c = threadIdx.x;
    float s = 0.0f;
    for (int r = 0; r < N; ++r) s += a[(size_t)r * DMODEL + c];
    pooled[c] = s / (float)N;
}

// ---------------------------------------------------------------------------
// fusion MLP: out = gelu(pooled@W1+b1)@W2+b2, single block of 256
// ---------------------------------------------------------------------------
__global__ void k_fusion(const float* __restrict__ pooled,
                         const float* __restrict__ W1, const float* __restrict__ b1,
                         const float* __restrict__ W2, const float* __restrict__ b2,
                         float* __restrict__ out) {
    __shared__ float hsh[DMODEL];
    int j = threadIdx.x;
    float acc = b1[j];
    for (int i = 0; i < 768; ++i) acc += pooled[i] * W1[(size_t)i * DMODEL + j];
    hsh[j] = gelu_tanh(acc);
    __syncthreads();
    float o = b2[j];
    for (int i = 0; i < DMODEL; ++i) o += hsh[i] * W2[(size_t)i * DMODEL + j];
    out[j] = o;
}

// ---------------------------------------------------------------------------
// host orchestration
// ---------------------------------------------------------------------------
static inline void launch_gemm(const float* A, const float* W, const float* b, float* C,
                               int M, int K, hipStream_t s) {
    k_gemm_bias<<<M / 16, 128, 0, s>>>(A, W, b, C, M, K);   // N fixed at 256
}

extern "C" void kernel_launch(void* const* d_in, const int* in_sizes, int n_in,
                              void* d_out, int out_size, void* d_ws, size_t ws_size,
                              hipStream_t stream) {
    (void)n_in; (void)out_size; (void)ws_size;

    const float* x0 = (const float*)d_in[0];
    const float* x1 = (const float*)d_in[1];
    const float* x2 = (const float*)d_in[2];
    const int* ei0 = (const int*)d_in[3];
    const int* ei1 = (const int*)d_in[4];
    const int* ei2 = (const int*)d_in[5];
    const float* ea0 = (const float*)d_in[6];
    const float* ea1 = (const float*)d_in[7];
    const float* ea2 = (const float*)d_in[8];

    int N0 = in_sizes[0] / 128, N1 = in_sizes[1] / 256, N2 = in_sizes[2] / 256;
    int E0 = in_sizes[6] / 32, E1 = in_sizes[7] / 32, E2 = in_sizes[8] / 32;

    // ---- workspace carve (256B aligned regions) ----
    char* ws = (char*)d_ws;
    size_t off = 0;
    auto carve = [&](size_t bytes) -> void* {
        void* p = ws + off;
        off += (bytes + 255) & ~(size_t)255;
        return p;
    };
    float* lout0 = (float*)carve((size_t)N0 * DMODEL * 4);
    float* lout1 = (float*)carve((size_t)N1 * DMODEL * 4);
    float* lout2 = (float*)carve((size_t)N2 * DMODEL * 4);
    float* a1    = (float*)carve((size_t)N1 * DMODEL * 4);
    float* a2    = (float*)carve((size_t)N2 * DMODEL * 4);
    float* hA    = (float*)carve((size_t)N0 * DMODEL * 4);
    float* hB    = (float*)carve((size_t)N0 * DMODEL * 4);
    float* qb    = (float*)carve((size_t)N0 * DMODEL * 4);
    float* kb    = (float*)carve((size_t)N0 * DMODEL * 4);
    float* vb    = (float*)carve((size_t)N0 * DMODEL * 4);
    float* agg   = (float*)carve((size_t)N0 * DMODEL * 4);
    float* logits = (float*)carve((size_t)E0 * HEADS * 4);
    unsigned* menc = (unsigned*)carve((size_t)N0 * HEADS * 4);
    float* sbuf  = (float*)carve((size_t)N0 * HEADS * 4);
    float* pooled = (float*)carve(768 * 4);

    auto P = [&](int i) -> const float* { return (const float*)d_in[i]; };

    // ---- per-level graph encoder ----
    // pbase: index of lvl.Wout; layer L params at base = pbase+2+L*12 in order
    // b,g, We,Wk,Wo,Wq,Wv, be,bk,bo,bq,bv (sorted pytree key order)
    auto run_level = [&](const float* x, int N, int in_dim,
                         const int* ei, const float* ea, int E,
                         int pbase, float* lout) {
        const float* cur = x;
        int nd = in_dim;
        int ethreads = E * HEADS;
        int eblocks = (ethreads + 255) / 256;
        for (int L = 0; L < 2; ++L) {
            int gb = pbase + 2 + L * 12;
            const float* bb = P(gb + 0);
            const float* gg = P(gb + 1);
            const float* We = P(gb + 2);
            const float* Wk = P(gb + 3);
            const float* Wo = P(gb + 4);
            const float* Wq = P(gb + 5);
            const float* Wv = P(gb + 6);
            const float* be = P(gb + 7);
            const float* bk = P(gb + 8);
            const float* bo = P(gb + 9);
            const float* bq = P(gb + 10);
            const float* bv = P(gb + 11);

            launch_gemm(cur, Wq, bq, qb, N, nd, stream);
            launch_gemm(cur, Wk, bk, kb, N, nd, stream);
            launch_gemm(cur, Wv, bv, vb, N, nd, stream);

            int zn1 = N * HEADS;
            int zn2 = N * DMODEL;
            k_fill_u32<<<(zn1 + 255) / 256, 256, 0, stream>>>(menc, 0u, zn1);
            k_fill_u32<<<(zn1 + 255) / 256, 256, 0, stream>>>((unsigned*)sbuf, 0u, zn1);
            k_fill_u32<<<(zn2 + 255) / 256, 256, 0, stream>>>((unsigned*)agg, 0u, zn2);

            k_edge_logits<<<eblocks, 256, 0, stream>>>(qb, kb, ea, We, be, ei, logits, E);
            k_seg_max<<<eblocks, 256, 0, stream>>>(logits, ei, menc, E);
            k_edge_exp<<<eblocks, 256, 0, stream>>>(logits, menc, ei, sbuf, E);
            k_edge_agg<<<eblocks, 256, 0, stream>>>(logits, sbuf, vb, ea, We, be, ei, agg, E);

            launch_gemm(agg, Wo, bo, hB, N, DMODEL, stream);
            k_ln_gelu<<<N, DMODEL, 0, stream>>>(hB, gg, bb, hA);
            cur = hA;
            nd = DMODEL;
        }
        launch_gemm(cur, P(pbase), P(pbase + 1), lout, N, DMODEL, stream);
    };

    run_level(x0, N0, 128, ei0, ea0, E0, 13, lout0);
    run_level(x1, N1, 256, ei1, ea1, E1, 39, lout1);
    run_level(x2, N2, 256, ei2, ea2, E2, 65, lout2);

    // ---- cross attention: pb = index of Wk; order Wk,Wo,Wq,Wv,bk,bo,bq,bv ----
    auto run_mha = [&](const float* qin, int Nq, const float* kvin, int Nk,
                       int pb, float* outb) {
        launch_gemm(qin, P(pb + 2), P(pb + 6), qb, Nq, DMODEL, stream);
        launch_gemm(kvin, P(pb + 0), P(pb + 4), kb, Nk, DMODEL, stream);
        launch_gemm(kvin, P(pb + 3), P(pb + 7), vb, Nk, DMODEL, stream);
        int t = Nq * HEADS;
        k_xattn<<<(t + 255) / 256, 256, 0, stream>>>(qb, kb, vb, hB, Nq, Nk);
        launch_gemm(hB, P(pb + 1), P(pb + 5), outb, Nq, DMODEL, stream);
    };
    run_mha(lout1, N1, lout0, N0, 91, a1);
    run_mha(lout2, N2, lout1, N1, 99, a2);

    // ---- pooling + fusion ----
    k_mean_pool<<<1, DMODEL, 0, stream>>>(lout0, pooled + 0, N0);
    k_mean_pool<<<1, DMODEL, 0, stream>>>(a1, pooled + 256, N1);
    k_mean_pool<<<1, DMODEL, 0, stream>>>(a2, pooled + 512, N2);
    k_fusion<<<1, DMODEL, 0, stream>>>(pooled, P(9), P(11), P(10), P(12), (float*)d_out);
}
